// DynamicRoutingAggregator_24799141167157
// MI455X (gfx1250) — matrix operand order repaired and checked by reading, not compiled, not run on hardware
//
#include <hip/hip_runtime.h>
#include <hip/hip_bf16.h>
#include <math.h>

// ---------------- problem constants ----------------
#define BATCH   32
#define TOKENS  2048
#define DIM     768
#define CAPS    16
#define CDIM    48
#define NOUT    (CAPS*CDIM)       // 768
#define MTOT    (BATCH*TOKENS)    // 65536
#define NCHUNK  16                // token chunks per batch (128 tokens each)

typedef __attribute__((ext_vector_type(16))) __bf16 v16bf;
typedef __attribute__((ext_vector_type(8)))  __bf16 v8bf;
typedef __attribute__((ext_vector_type(8)))  float  v8f;

union ABfrag { v16bf v; v8bf h[2]; };

static __device__ __forceinline__ float bf16lo_to_f(unsigned u) {
    union { unsigned x; float f; } c; c.x = u << 16;        return c.f;
}
static __device__ __forceinline__ float bf16hi_to_f(unsigned u) {
    union { unsigned x; float f; } c; c.x = u & 0xffff0000u; return c.f;
}

// =====================================================================
// Kernel 1: shared = tanh(X * W^T + b), stored as bf16 (fits 192MB L2).
// Block tile 128(M) x 64(N), K-step 32, 8 waves -> each wave 32x32
// (four 16x16 bf16 WMMA accumulators).
// =====================================================================
__global__ __launch_bounds__(256)
void gemm_tanh_bf16(const float* __restrict__ X, const float* __restrict__ W,
                    const float* __restrict__ bias, __bf16* __restrict__ S)
{
    __shared__ __align__(16) __bf16 Alds[128][32];  // 8KB  (X tile, bf16)
    __shared__ __align__(16) __bf16 Blds[64][32];   // 4KB  (W tile, bf16; row = out-feature)

    const int tid  = threadIdx.x;
    const int lane = tid & 31;
    const int w    = tid >> 5;            // wave 0..7
    const int wm   = (w & 3) * 32;        // wave M offset in block tile
    const int wn   = (w >> 2) * 32;       // wave N offset in block tile
    const int gN   = blockIdx.x;          // 0..11  (innermost -> L2 reuse of X)
    const int gM   = blockIdx.y;          // 0..511

    v8f acc[2][2];
    #pragma unroll
    for (int i = 0; i < 2; ++i)
        #pragma unroll
        for (int j = 0; j < 2; ++j)
            acc[i][j] = (v8f){0.f,0.f,0.f,0.f,0.f,0.f,0.f,0.f};

    const int seg  = tid & 7;             // 4-float segment within 32-col row
    const int row0 = tid >> 3;            // 0..31

    for (int k0 = 0; k0 < DIM; k0 += 32) {
        // --- stage X 128x32 f32 -> bf16 LDS ---
        #pragma unroll
        for (int p = 0; p < 4; ++p) {
            const int r = row0 + p * 32;
            const float4 x = *(const float4*)&X[(size_t)(gM*128 + r)*DIM + k0 + seg*4];
            union { __bf16 h[4]; uint2 u; } pk;
            pk.h[0] = (__bf16)x.x; pk.h[1] = (__bf16)x.y;
            pk.h[2] = (__bf16)x.z; pk.h[3] = (__bf16)x.w;
            *(uint2*)&Alds[r][seg*4] = pk.u;
        }
        // --- stage W 64x32 f32 -> bf16 LDS ---
        #pragma unroll
        for (int p = 0; p < 2; ++p) {
            const int r = row0 + p * 32;
            const float4 x = *(const float4*)&W[(size_t)(gN*64 + r)*DIM + k0 + seg*4];
            union { __bf16 h[4]; uint2 u; } pk;
            pk.h[0] = (__bf16)x.x; pk.h[1] = (__bf16)x.y;
            pk.h[2] = (__bf16)x.z; pk.h[3] = (__bf16)x.w;
            *(uint2*)&Blds[r][seg*4] = pk.u;
        }
        __syncthreads();

        // --- build fragments per ISA bf16 layouts ---
        // A 16x32: lane L row M=L&15; elems 0..7 = K[klo..klo+7], 8..15 = K[klo+16..klo+23], klo=(L>=16)?8:0
        // B 32x16: lane L col N=L&15; elems 0..7 = K[kb..kb+7], 8..15 = K[kb+16..kb+23], kb=(L>=16)?8:0  (B col n = W row n)
        const int mrow = lane & 15;
        const int klo  = (lane >> 4) * 8;
        const int kb   = (lane >> 4) * 8;
        ABfrag a[2], bfr[2];
        #pragma unroll
        for (int i = 0; i < 2; ++i) {
            const int r = wm + i*16 + mrow;
            a[i].h[0] = *(const v8bf*)&Alds[r][klo];
            a[i].h[1] = *(const v8bf*)&Alds[r][klo + 16];
        }
        #pragma unroll
        for (int j = 0; j < 2; ++j) {
            const int n = wn + j*16 + mrow;
            bfr[j].h[0] = *(const v8bf*)&Blds[n][kb];
            bfr[j].h[1] = *(const v8bf*)&Blds[n][kb + 16];
        }

        #pragma unroll
        for (int i = 0; i < 2; ++i)
            #pragma unroll
            for (int j = 0; j < 2; ++j)
                acc[i][j] = __builtin_amdgcn_wmma_f32_16x16x32_bf16(
                    false, a[i].v, false, bfr[j].v, (short)0, acc[i][j], false, false);

        __syncthreads();
    }

    // --- epilogue: bias + tanh, store bf16 ---
    // C/D layout: VGPR e, lanes 0-15 -> M=e, lanes 16-31 -> M=e+8; N = lane&15
    const int nlane = lane & 15;
    const int mhi   = (lane >> 4) * 8;
    #pragma unroll
    for (int j = 0; j < 2; ++j) {
        const int col = gN*64 + wn + j*16 + nlane;
        const float bj = bias[col];
        #pragma unroll
        for (int i = 0; i < 2; ++i) {
            #pragma unroll
            for (int e = 0; e < 8; ++e) {
                const int rowg = gM*128 + wm + i*16 + mhi + e;
                const float v = tanhf(acc[i][j][e] + bj);
                S[(size_t)rowg * NOUT + col] = (__bf16)v;
            }
        }
    }
}

// =====================================================================
// Kernel 2 (fused routing step, one pass over `shared`):
//   B[t,c] += dot(shared[t,c,:], V[c,:]);  C = softmax_caps(B)*mask;
//   P[b,chunk,:] += C[t,c]*shared[t,c,:]
// lane = capsule (16 lanes/half-wave), softmax via 16-wide shfl_xor tree.
// With V==0 && B==0 this reproduces iteration 0 exactly (uniform 1/16).
// =====================================================================
__global__ __launch_bounds__(256)
void route_update(const __bf16* __restrict__ S, const int* __restrict__ mask,
                  const float* __restrict__ Vin, float* __restrict__ Bmat,
                  float* __restrict__ P)
{
    const int chunk = blockIdx.x;    // 0..15
    const int b     = blockIdx.y;    // 0..31
    const int tid   = threadIdx.x;
    const int slot  = tid >> 4;      // 0..15 token slots
    const int cap   = tid & 15;      // capsule = lane within half-wave

    __shared__ float accl[16][NOUT]; // 48KB partials per slot

    // V for this lane's capsule (fixed per thread)
    float Vreg[CDIM];
    #pragma unroll
    for (int d = 0; d < CDIM; ++d) Vreg[d] = Vin[b*NOUT + cap*CDIM + d];

    float acc[CDIM];
    #pragma unroll
    for (int d = 0; d < CDIM; ++d) acc[d] = 0.f;

    for (int it = 0; it < 8; ++it) {
        const int t = chunk*128 + it*16 + slot;
        const int m = mask[b*TOKENS + t];
        if (m != 0) {
            const size_t base = (size_t)(b*TOKENS + t) * NOUT + cap*CDIM;
            const uint4* sp4 = (const uint4*)(S + base);  // 48 bf16 = 6 x uint4
            float sv[CDIM];
            #pragma unroll
            for (int q = 0; q < 6; ++q) {
                const uint4 u = sp4[q];
                sv[8*q+0] = bf16lo_to_f(u.x); sv[8*q+1] = bf16hi_to_f(u.x);
                sv[8*q+2] = bf16lo_to_f(u.y); sv[8*q+3] = bf16hi_to_f(u.y);
                sv[8*q+4] = bf16lo_to_f(u.z); sv[8*q+5] = bf16hi_to_f(u.z);
                sv[8*q+6] = bf16lo_to_f(u.w); sv[8*q+7] = bf16hi_to_f(u.w);
            }
            float dot = 0.f;
            #pragma unroll
            for (int d = 0; d < CDIM; ++d) dot = fmaf(sv[d], Vreg[d], dot);

            const size_t bidx = (size_t)(b*TOKENS + t)*CAPS + cap;
            const float Bv = Bmat[bidx] + dot;
            Bmat[bidx] = Bv;

            // softmax over 16 capsules (16-lane xor tree, wave32 half-wave)
            float mx = Bv;
            mx = fmaxf(mx, __shfl_xor(mx, 1, 16));
            mx = fmaxf(mx, __shfl_xor(mx, 2, 16));
            mx = fmaxf(mx, __shfl_xor(mx, 4, 16));
            mx = fmaxf(mx, __shfl_xor(mx, 8, 16));
            float e = __expf(Bv - mx);
            float s = e;
            s += __shfl_xor(s, 1, 16);
            s += __shfl_xor(s, 2, 16);
            s += __shfl_xor(s, 4, 16);
            s += __shfl_xor(s, 8, 16);
            const float C = e / s;   // mask==1 here; masked tokens contribute 0

            #pragma unroll
            for (int d = 0; d < CDIM; ++d) acc[d] = fmaf(C, sv[d], acc[d]);
        }
    }

    // every (slot,cap,d) cell written exactly once -> no zeroing needed
    #pragma unroll
    for (int d = 0; d < CDIM; ++d) accl[slot][cap*CDIM + d] = acc[d];
    __syncthreads();

    for (int k = tid; k < NOUT; k += 256) {
        float s = 0.f;
        #pragma unroll
        for (int sl = 0; sl < 16; ++sl) s += accl[sl][k];
        P[((size_t)b*NCHUNK + chunk)*NOUT + k] = s;   // deterministic reduction
    }
}

// =====================================================================
// Kernel 3: S = sum_chunks P;  V = squash(S) along cdim=48
// =====================================================================
__global__ __launch_bounds__(768)
void reduce_squash(const float* __restrict__ P, float* __restrict__ Vout)
{
    const int b = blockIdx.x;
    const int k = threadIdx.x;       // 0..767
    __shared__ float sm[NOUT];

    float s = 0.f;
    #pragma unroll
    for (int c = 0; c < NCHUNK; ++c) s += P[((size_t)b*NCHUNK + c)*NOUT + k];
    sm[k] = s;
    __syncthreads();

    const int base = (k / CDIM) * CDIM;
    float n = 0.f;
    #pragma unroll
    for (int d = 0; d < CDIM; ++d) { const float x = sm[base + d]; n = fmaf(x, x, n); }
    const float v = s * (n / (1.f + n)) * rsqrtf(n + 1e-8f);
    Vout[b*NOUT + k] = v;
}

// =====================================================================
extern "C" void kernel_launch(void* const* d_in, const int* in_sizes, int n_in,
                              void* d_out, int out_size, void* d_ws, size_t ws_size,
                              hipStream_t stream)
{
    const float* X    = (const float*)d_in[0];  // (32,2048,768) f32
    const int*   mask = (const int*)  d_in[1];  // (32,2048) i32
    const float* W    = (const float*)d_in[2];  // (768,768) f32
    const float* bias = (const float*)d_in[3];  // (768,)    f32
    float* out = (float*)d_out;                 // (32,768)  f32

    char* ws = (char*)d_ws;
    size_t off = 0;
    __bf16* Sh  = (__bf16*)(ws + off); off += (size_t)MTOT*NOUT*sizeof(__bf16);        // 100.7MB (L2-resident)
    float*  Bm  = (float*) (ws + off); off += (size_t)MTOT*CAPS*sizeof(float);         // 4MB
    float*  P   = (float*) (ws + off); off += (size_t)BATCH*NCHUNK*NOUT*sizeof(float); // 1.5MB
    float*  V   = (float*) (ws + off); off += (size_t)BATCH*NOUT*sizeof(float);        // 96KB

    hipMemsetAsync(Bm, 0, (size_t)MTOT*CAPS*sizeof(float), stream);
    hipMemsetAsync(V,  0, (size_t)BATCH*NOUT*sizeof(float), stream);

    // GEMM+tanh: N-tiles innermost for L2 reuse of X rows
    gemm_tanh_bf16<<<dim3(NOUT/64, MTOT/128), 256, 0, stream>>>(X, W, bias, Sh);

    for (int it = 0; it < 3; ++it) {
        route_update<<<dim3(NCHUNK, BATCH), 256, 0, stream>>>(Sh, mask, V, Bm, P);
        reduce_squash<<<BATCH, NOUT, 0, stream>>>(P, (it == 2) ? out : V);
    }
}
